// TransformerEncoderLayerV2_90503550861972
// MI455X (gfx1250) — matrix operand order repaired and checked by reading, not compiled, not run on hardware
//
#include <hip/hip_runtime.h>

// ---------------------------------------------------------------------------
// Problem constants (from the reference)
// ---------------------------------------------------------------------------
#define Bq      32
#define Mseq    399
#define Dm      512
#define Hh      8
#define HD      64          // head dim
#define FFd     2048
#define CLIPc   12
#define STRIDEc 9           // CLIP - OVERLAP
#define Cc      44          // (M - CLIP)/STRIDE + 1
#define PADc    16
#define CPD     (PADc * Dm) // 8192

// CDNA5 async global->LDS staging (ASYNCcnt path); assembled OK in round 2.
#define USE_ASYNC_LDS 1

typedef __attribute__((ext_vector_type(16))) _Float16 v16h;
typedef __attribute__((ext_vector_type(8)))  float    v8f;

// ---------------------------------------------------------------------------
// WMMA helpers (CDNA5 wave32, V_WMMA_F32_16X16X32_F16)
// A 16x32 f16 fragment: lane&15 = M row; lane>>4 selects K 0..15 vs
// 16..31 (contiguous, as B); pairs packed per VGPR (ISA 7.12.2).
// B 32x16 f16 fragment: lane&15 = N col; lanes 0-15 hold K=0..15 linear,
// lanes 16-31 hold K=16..31.
// C/D 16x16 f32: lane&15 = N; VGPR v -> M = v + 8*(lane>>4).
// ---------------------------------------------------------------------------
__device__ __forceinline__ v8f wmma32(v16h a, v16h b, v8f c) {
  return __builtin_amdgcn_wmma_f32_16x16x32_f16(
      /*neg_a=*/false, a, /*neg_b=*/false, b,
      /*c_mod=*/(short)0, c, /*reuse_a=*/false, /*reuse_b=*/false);
}

// A fragment from row-major (M x K) LDS tile; one contiguous 16-half run.
__device__ __forceinline__ v16h frag_a_lds(const _Float16* lds, int ld, int r0, int k0) {
  const int lane = threadIdx.x & 31;
  const _Float16* p = lds + (size_t)(r0 + (lane & 15)) * ld + k0 + ((lane >> 4) << 4);
  v16h a;
#pragma unroll
  for (int vv = 0; vv < 8; ++vv) {
    int koff = 2 * vv;
    a[2 * vv]     = p[koff];
    a[2 * vv + 1] = p[koff + 1];
  }
  return a;
}

// B fragment where logical B[kk][n] = lds[n*ld + k0 + kk]; i.e. the LDS tile
// holds B transposed (N x K) -> one contiguous 16-half (2 x b128) run.
__device__ __forceinline__ v16h frag_bt_lds(const _Float16* lds, int ld, int k0, int n0) {
  const int lane = threadIdx.x & 31;
  const _Float16* p = lds + (size_t)(n0 + (lane & 15)) * ld + k0 + ((lane >> 4) << 4);
  v16h b;
#pragma unroll
  for (int vv = 0; vv < 16; ++vv) b[vv] = p[vv];
  return b;
}

// ---------------------------------------------------------------------------
// Elementwise / reduction kernels
// ---------------------------------------------------------------------------

// 32x32 tiled transpose + f32->f16 convert: in (K x N) f32 -> out (N x K) f16.
__global__ __launch_bounds__(256) void transpose_f32_to_f16(
    const float* __restrict__ in, _Float16* __restrict__ out, int K, int N) {
  __shared__ _Float16 t[32][33];
  const int k0 = blockIdx.y * 32, n0 = blockIdx.x * 32;
  const int tx = threadIdx.x & 31, ty = threadIdx.x >> 5;  // 32 x 8
  for (int r = ty; r < 32; r += 8)
    t[r][tx] = (_Float16)in[(size_t)(k0 + r) * N + n0 + tx];
  __syncthreads();
  for (int r = ty; r < 32; r += 8)
    out[(size_t)(n0 + r) * K + k0 + tx] = t[tx][r];
}

// One block per row; LayerNorm over W columns, output f16 (GEMM input)
__global__ __launch_bounds__(256) void layernorm_to_h(
    const float* __restrict__ x, const float* __restrict__ g,
    const float* __restrict__ bb, _Float16* __restrict__ out, int W) {
  const int row = blockIdx.x;
  const float* xr = x + (size_t)row * W;
  __shared__ float red[256];
  float s = 0.f;
  for (int i = threadIdx.x; i < W; i += 256) s += xr[i];
  red[threadIdx.x] = s; __syncthreads();
  for (int o = 128; o > 0; o >>= 1) {
    if (threadIdx.x < o) red[threadIdx.x] += red[threadIdx.x + o];
    __syncthreads();
  }
  const float mean = red[0] / (float)W;
  __syncthreads();
  float s2 = 0.f;
  for (int i = threadIdx.x; i < W; i += 256) { float d = xr[i] - mean; s2 += d * d; }
  red[threadIdx.x] = s2; __syncthreads();
  for (int o = 128; o > 0; o >>= 1) {
    if (threadIdx.x < o) red[threadIdx.x] += red[threadIdx.x + o];
    __syncthreads();
  }
  const float rstd = rsqrtf(red[0] / (float)W + 1e-6f);
  _Float16* orow = out + (size_t)row * W;
  for (int i = threadIdx.x; i < W; i += 256)
    orow[i] = (_Float16)((xr[i] - mean) * rstd * g[i] + bb[i]);
}

// Gather clips + boundary-extension adds + zero-pad -> flat (B*C, PAD*D)
__global__ __launch_bounds__(256) void gather_clips(const float* __restrict__ x,
                                                    float* __restrict__ flat) {
  const int bc = blockIdx.x;
  const int b = bc / Cc, c = bc % Cc;
  const float* xb = x + (size_t)b * Mseq * Dm;
  float* out = flat + (size_t)bc * CPD;
  const bool mid = (c >= 1) && (c <= Cc - 2);
  for (int d = threadIdx.x; d < Dm; d += 256) {
    float df = 0.f, dl = 0.f;
#pragma unroll
    for (int k = 1; k <= 2; ++k) {  // EXT/2 = 2
      if (mid && c >= k)          df += xb[(size_t)((c - k) * STRIDEc) * Dm + d];
      if (mid && c + k <= Cc - 1) dl += xb[(size_t)((c + k) * STRIDEc + CLIPc - 1) * Dm + d];
    }
    const int nE = 4;  // min(EXT, C-1)
    if (c == 0)
      for (int t = 1; t <= nE; ++t) dl += xb[(size_t)(t * STRIDEc + CLIPc - 1) * Dm + d];
    if (c == Cc - 1)
      for (int t = Cc - 1 - nE; t <= Cc - 2; ++t) df += xb[(size_t)(t * STRIDEc) * Dm + d];
    for (int j = 0; j < PADc; ++j) {
      float v = 0.f;
      if (j < CLIPc) {
        v = xb[(size_t)(c * STRIDEc + j) * Dm + d];
        if (j == 0)         v += df;
        if (j == CLIPc - 1) v += dl;
      }
      out[(size_t)j * Dm + d] = v;
    }
  }
}

// x2 = x + overlap-scatter of rev; per-output gather over (<=2) clips.
__global__ void combine_x2(const float* __restrict__ x, const float* __restrict__ rev,
                           float* __restrict__ x2) {
  const int idx = blockIdx.x * 256 + threadIdx.x;
  const int total = Bq * Mseq * Dm;
  if (idx >= total) return;
  const int d = idx % Dm;
  const int bm = idx / Dm;
  const int m = bm % Mseq, b = bm / Mseq;
  int chi = m / STRIDEc;            if (chi > Cc - 1) chi = Cc - 1;
  int clo = (m - (CLIPc - 1) + (STRIDEc - 1)) / STRIDEc;  if (clo < 0) clo = 0;
  float acc = 0.f;
  for (int c = clo; c <= chi; ++c) {
    const int j = m - c * STRIDEc;  // in [0, CLIP-1]
    acc += rev[(((size_t)b * Cc + c) * PADc + j) * Dm + d];
  }
  x2[idx] = x[idx] + acc;
}

// ---------------------------------------------------------------------------
// Double-buffered WMMA GEMM: C = A(f16, MxK) @ B + bias, B pre-transposed as
// Bt (N x K). Block tile 64x128, BK=64, 8 waves. Tile t+1 is staged via async
// global->LDS while tile t computes; in-order ASYNCcnt completion lets us wait
// on "<= ops-in-flight-for-next-tile" to know the current tile landed.
// Per wave per tile: B = 4 async b128, A = 2 async b128 (full blocks only;
// ragged tail block uses guarded sync stores so issue counts stay uniform).
// EPI: 0 f32=val | 1 f16=relu(val) | 2 f32=val+res
//      3 f16 attention layout (B,H,S,64), val*scale
//      5 f32=val & f16=val | 6 f32=f16=val+res
// ---------------------------------------------------------------------------
#define ALD 72   // As row stride (halfs): 144B, 16B-aligned, spreads banks
#define BLD 72   // Bs row stride (halfs)

template <int EPI>
__global__ __launch_bounds__(256) void gemm_kernel(
    const _Float16* __restrict__ A, const _Float16* __restrict__ Bt,
    const float* __restrict__ bias, const float* __restrict__ resid,
    float* __restrict__ outF, _Float16* __restrict__ outH,
    int Mrows, int N, int K, int S, float scale) {
  __shared__ _Float16 As[2][64 * ALD];
  __shared__ _Float16 Bs[2][128 * BLD];
  const int tid = threadIdx.x, w = tid >> 5, lane = tid & 31;
  const int rb = w >> 1;            // 0..3 row block
  const int cbB = (w & 1) * 4;      // first of 4 col blocks
  const int m0 = blockIdx.y * 64, n0 = blockIdx.x * 128;
  v8f acc[4] = {{}, {}, {}, {}};
  const int arow = tid >> 2, acol = (tid & 3) * 16;   // A: 16 halfs/thread (64x64)
  const int brow = tid >> 1, bcol = (tid & 1) * 32;   // Bt: 32 halfs/thread (128x64)
  const bool fullA = (m0 + 64 <= Mrows);  // block-uniform -> uniform ASYNCcnt

  auto stageB = [&](int buf, int k0) {
    const _Float16* src = Bt + (size_t)(n0 + brow) * K + k0 + bcol;
#if USE_ASYNC_LDS
    const unsigned lds0 = (unsigned)(size_t)&Bs[buf][brow * BLD + bcol];
    const unsigned long long ga = (unsigned long long)(size_t)src;
#pragma unroll
    for (int i = 0; i < 4; ++i)
      asm volatile("global_load_async_to_lds_b128 %0, %1, off"
                   :: "v"(lds0 + i * 16), "v"(ga + i * 16) : "memory");
#else
#pragma unroll
    for (int i = 0; i < 32; ++i) Bs[buf][brow * BLD + bcol + i] = src[i];
#endif
  };
  auto stageA = [&](int buf, int k0) {
#if USE_ASYNC_LDS
    if (fullA) {
      const _Float16* src = A + (size_t)(m0 + arow) * K + k0 + acol;
      const unsigned lds0 = (unsigned)(size_t)&As[buf][arow * ALD + acol];
      const unsigned long long ga = (unsigned long long)(size_t)src;
#pragma unroll
      for (int i = 0; i < 2; ++i)
        asm volatile("global_load_async_to_lds_b128 %0, %1, off"
                     :: "v"(lds0 + i * 16), "v"(ga + i * 16) : "memory");
      return;
    }
#endif
    const int gr = m0 + arow;
    if (gr < Mrows) {
      const _Float16* src = A + (size_t)gr * K + k0 + acol;
#pragma unroll
      for (int i = 0; i < 16; ++i) As[buf][arow * ALD + acol + i] = src[i];
    } else {
#pragma unroll
      for (int i = 0; i < 16; ++i) As[buf][arow * ALD + acol + i] = (_Float16)0.f;
    }
  };

  const int nt = K >> 6;       // K/64 tiles; K%64==0 at all call sites
  stageB(0, 0);
  stageA(0, 0);
  for (int t = 0; t < nt; ++t) {
    const int cur = t & 1, nxt = cur ^ 1;
    const int k0 = t << 6;
    if (t + 1 < nt) {
      if (k0 + 128 < K)  // L2 prefetch two tiles ahead (global_prefetch_b8)
        __builtin_prefetch(Bt + (size_t)(n0 + brow) * K + k0 + 128 + bcol, 0, 1);
      stageB(nxt, k0 + 64);    // in flight during compute of tile t
      stageA(nxt, k0 + 64);
#if USE_ASYNC_LDS
      if (fullA) asm volatile("s_wait_asynccnt 0x6" ::: "memory");  // 4B+2A next
      else       asm volatile("s_wait_asynccnt 0x4" ::: "memory");  // 4B next
#endif
    } else {
#if USE_ASYNC_LDS
      asm volatile("s_wait_asynccnt 0x0" ::: "memory");
#endif
    }
    __syncthreads();           // current tile visible to all waves
#pragma unroll
    for (int ks = 0; ks < 64; ks += 32) {
      const v16h af = frag_a_lds(&As[cur][0], ALD, rb * 16, ks);
#pragma unroll
      for (int j = 0; j < 4; ++j) {
        const v16h bf = frag_bt_lds(&Bs[cur][0], BLD, ks, (cbB + j) * 16);
        acc[j] = wmma32(af, bf, acc[j]);
      }
    }
    __syncthreads();           // all reads of `cur` done before it is restaged
  }

#pragma unroll
  for (int j = 0; j < 4; ++j) {
    const int col = n0 + (cbB + j) * 16 + (lane & 15);
#pragma unroll
    for (int v = 0; v < 8; ++v) {
      const int row = m0 + rb * 16 + v + ((lane >> 4) << 3);
      if (row < Mrows) {
        float val = acc[j][v] + bias[col];
        if constexpr (EPI == 0) {
          outF[(size_t)row * N + col] = val;
        } else if constexpr (EPI == 1) {
          outH[(size_t)row * N + col] = (_Float16)fmaxf(val, 0.f);
        } else if constexpr (EPI == 2) {
          outF[(size_t)row * N + col] = val + resid[(size_t)row * N + col];
        } else if constexpr (EPI == 3) {
          const int b = row / S, s = row - b * S, h = col >> 6, d = col & 63;
          outH[(((size_t)b * Hh + h) * S + s) * HD + d] = (_Float16)(val * scale);
        } else if constexpr (EPI == 5) {
          outF[(size_t)row * N + col] = val;
          outH[(size_t)row * N + col] = (_Float16)val;
        } else if constexpr (EPI == 6) {
          const float r2 = val + resid[(size_t)row * N + col];
          outF[(size_t)row * N + col] = r2;
          outH[(size_t)row * N + col] = (_Float16)r2;
        }
      }
    }
  }
}

// ---------------------------------------------------------------------------
// Flash attention. q/k/v f16 in (B*H, S, 64); q pre-scaled by 1/sqrt(hd).
// 64 q-rows per block, 32 kv per step; S=Q.K^T and P.V via WMMA; online
// softmax in LDS; O fragments in registers; V staged transposed so the P.V
// B-fragments are contiguous. grid=(qtiles, B*H).
// ---------------------------------------------------------------------------
#define QLD 72
#define KLD 72
#define VLD 40   // Vt row stride (halfs)
#define PLD 40

__global__ __launch_bounds__(256) void flash_attn(
    const _Float16* __restrict__ q, const _Float16* __restrict__ k,
    const _Float16* __restrict__ v, _Float16* __restrict__ ctx, int S) {
  const int qt0 = blockIdx.x * 64;
  const int bh = blockIdx.y;
  const int b = bh / Hh, h = bh % Hh;
  const _Float16* qp = q + (size_t)bh * S * HD;
  const _Float16* kp = k + (size_t)bh * S * HD;
  const _Float16* vp = v + (size_t)bh * S * HD;

  __shared__ _Float16 Qs[64 * QLD];
  __shared__ _Float16 Ks[32 * KLD];   // (kv, hd) row-major
  __shared__ _Float16 Vt[64 * VLD];   // (hd, kv) transposed
  __shared__ float    Ss[64 * 33];
  __shared__ _Float16 Ps[64 * PLD];
  __shared__ float mrow[64], lrow[64], srow[64];

  const int tid = threadIdx.x, w = tid >> 5, lane = tid & 31;
  const int rb  = w >> 1;        // q row-block (0..3)
  const int cbk = w & 1;         // score col-block (0..1)
  const int ob0 = (w & 1) * 2;   // O col-blocks {ob0, ob0+1} of hd

  {  // load Q tile (zero-fill beyond S)
    const int qrow = tid >> 2, qcol = (tid & 3) * 16;
    const int gr = qt0 + qrow;
    if (gr < S) {
      const _Float16* src = qp + (size_t)gr * HD + qcol;
#pragma unroll
      for (int i = 0; i < 16; ++i) Qs[qrow * QLD + qcol + i] = src[i];
    } else {
#pragma unroll
      for (int i = 0; i < 16; ++i) Qs[qrow * QLD + qcol + i] = (_Float16)0.f;
    }
  }
  if (tid < 64) { mrow[tid] = -3.0e38f; lrow[tid] = 0.f; }
  v8f o0 = {}, o1 = {};
  __syncthreads();

  for (int kv0 = 0; kv0 < S; kv0 += 32) {
    {  // stage K row-major and V transposed
      const int krow = tid >> 3, kcol = (tid & 7) * 8;
      if (kv0 + krow < S) {
        const _Float16* ks = kp + (size_t)(kv0 + krow) * HD + kcol;
        const _Float16* vs = vp + (size_t)(kv0 + krow) * HD + kcol;
#pragma unroll
        for (int i = 0; i < 8; ++i) {
          Ks[krow * KLD + kcol + i] = ks[i];
          Vt[(kcol + i) * VLD + krow] = vs[i];
        }
      } else {
#pragma unroll
        for (int i = 0; i < 8; ++i) {
          Ks[krow * KLD + kcol + i] = (_Float16)0.f;
          Vt[(kcol + i) * VLD + krow] = (_Float16)0.f;
        }
      }
    }
    __syncthreads();

    // S = Q @ K^T : each wave one 16x16 fragment; hd=64 -> 2 WMMA k-steps
    v8f sacc = {};
#pragma unroll
    for (int ks = 0; ks < 64; ks += 32) {
      const v16h af = frag_a_lds(Qs, QLD, rb * 16, ks);
      const v16h bf = frag_bt_lds(Ks, KLD, ks, cbk * 16);  // B[kk][n]=K[n][kk]
      sacc = wmma32(af, bf, sacc);
    }
#pragma unroll
    for (int vv = 0; vv < 8; ++vv) {
      const int r = rb * 16 + vv + ((lane >> 4) << 3);
      Ss[r * 33 + cbk * 16 + (lane & 15)] = sacc[vv];
    }
    __syncthreads();

    // online softmax (one thread per q row)
    if (tid < 64) {
      const int r = tid;
      const int lim = S - kv0;
      float mx = mrow[r];
      for (int j = 0; j < 32; ++j)
        if (j < lim) mx = fmaxf(mx, Ss[r * 33 + j]);
      const float sc = __expf(mrow[r] - mx);
      float ls = lrow[r] * sc;
      for (int j = 0; j < 32; ++j) {
        const float p = (j < lim) ? __expf(Ss[r * 33 + j] - mx) : 0.f;
        Ps[r * PLD + j] = (_Float16)p;
        ls += p;
      }
      mrow[r] = mx; lrow[r] = ls; srow[r] = sc;
    }
    __syncthreads();

    // O = O*scale + P @ V
#pragma unroll
    for (int vv = 0; vv < 8; ++vv) {
      const int r = rb * 16 + vv + ((lane >> 4) << 3);
      const float sc = srow[r];
      o0[vv] *= sc; o1[vv] *= sc;
    }
    const v16h pf = frag_a_lds(Ps, PLD, rb * 16, 0);
    const v16h bf0 = frag_bt_lds(Vt, VLD, 0, (ob0 + 0) * 16);  // B[kk][n]=V[kk][n]
    o0 = wmma32(pf, bf0, o0);
    const v16h bf1 = frag_bt_lds(Vt, VLD, 0, (ob0 + 1) * 16);
    o1 = wmma32(pf, bf1, o1);
    __syncthreads();
  }

  // normalize + store ctx (f16, (B*S, D) with cols h*64+d)
#pragma unroll
  for (int vv = 0; vv < 8; ++vv) {
    const int rl = rb * 16 + vv + ((lane >> 4) << 3);
    const int gq = qt0 + rl;
    if (gq < S) {
      const float inv = 1.f / lrow[rl];
      const int c0 = (ob0 + 0) * 16 + (lane & 15);
      const int c1 = (ob0 + 1) * 16 + (lane & 15);
      ctx[((size_t)b * S + gq) * Dm + h * HD + c0] = (_Float16)(o0[vv] * inv);
      ctx[((size_t)b * S + gq) * Dm + h * HD + c1] = (_Float16)(o1[vv] * inv);
    }
  }
}

// ---------------------------------------------------------------------------
// Host-side orchestration
// ---------------------------------------------------------------------------
extern "C" void kernel_launch(void* const* d_in, const int* in_sizes, int n_in,
                              void* d_out, int out_size, void* d_ws, size_t ws_size,
                              hipStream_t stream) {
  (void)in_sizes; (void)n_in; (void)out_size; (void)ws_size;
  const float* x       = (const float*)d_in[0];
  const float* ln_g    = (const float*)d_in[2];
  const float* ln_b    = (const float*)d_in[3];
  const float* Wq      = (const float*)d_in[4];   const float* bqv  = (const float*)d_in[5];
  const float* Wk      = (const float*)d_in[6];   const float* bkv  = (const float*)d_in[7];
  const float* Wv      = (const float*)d_in[8];   const float* bvv  = (const float*)d_in[9];
  const float* Wo      = (const float*)d_in[10];  const float* bov  = (const float*)d_in[11];
  const float* ff_ln_g = (const float*)d_in[12];  const float* ff_ln_b = (const float*)d_in[13];
  const float* ff_W1   = (const float*)d_in[14];  const float* ff_b1   = (const float*)d_in[15];
  const float* ff_W2   = (const float*)d_in[16];  const float* ff_b2   = (const float*)d_in[17];
  const float* cl_ln_g = (const float*)d_in[18];  const float* cl_ln_b = (const float*)d_in[19];
  const float* cl_W    = (const float*)d_in[20];  const float* cl_b    = (const float*)d_in[21];
  const float* cWq     = (const float*)d_in[22];  const float* cbq     = (const float*)d_in[23];
  const float* cWk     = (const float*)d_in[24];  const float* cbk     = (const float*)d_in[25];
  const float* cWv     = (const float*)d_in[26];  const float* cbv     = (const float*)d_in[27];
  const float* cWo     = (const float*)d_in[28];  const float* cbo     = (const float*)d_in[29];
  const float* cff_ln_g= (const float*)d_in[30];  const float* cff_ln_b= (const float*)d_in[31];
  const float* cff_W1  = (const float*)d_in[32];  const float* cff_b1  = (const float*)d_in[33];
  const float* cff_W2  = (const float*)d_in[34];  const float* cff_b2  = (const float*)d_in[35];
  const float* rev_W   = (const float*)d_in[36];  const float* rev_b   = (const float*)d_in[37];

  const int RC = Bq * Cc;     // 1408 clip rows
  const int RM = Bq * Mseq;   // 12768 main rows

  // bump allocator over d_ws (~303 MB total)
  size_t off = 0;
  auto a16 = [&](size_t n) { _Float16* p = (_Float16*)((char*)d_ws + off);
                             off += ((n * 2 + 255) & ~(size_t)255); return p; };
  auto a32 = [&](size_t n) { float* p = (float*)((char*)d_ws + off);
                             off += ((n * 4 + 255) & ~(size_t)255); return p; };

  // weights, pre-transposed to (N x K) f16
  _Float16* clW_t  = a16((size_t)CPD * Dm);
  _Float16* cWq_t  = a16((size_t)Dm * Dm);
  _Float16* cWk_t  = a16((size_t)Dm * Dm);
  _Float16* cWv_t  = a16((size_t)Dm * Dm);
  _Float16* cWo_t  = a16((size_t)Dm * Dm);
  _Float16* cW1_t  = a16((size_t)Dm * FFd);
  _Float16* cW2_t  = a16((size_t)FFd * Dm);
  _Float16* revW_t = a16((size_t)Dm * CPD);
  _Float16* Wq_t   = a16((size_t)Dm * Dm);
  _Float16* Wk_t   = a16((size_t)Dm * Dm);
  _Float16* Wv_t   = a16((size_t)Dm * Dm);
  _Float16* Wo_t   = a16((size_t)Dm * Dm);
  _Float16* W1_t   = a16((size_t)Dm * FFd);
  _Float16* W2_t   = a16((size_t)FFd * Dm);

  float*    flat  = a32((size_t)RC * CPD);   // reused later as `rev`
  _Float16* flatn = a16((size_t)RC * CPD);
  float*    cl    = a32((size_t)RC * Dm);
  _Float16* cl_h  = a16((size_t)RC * Dm);
  _Float16* qc    = a16((size_t)RC * Dm);
  _Float16* kc    = a16((size_t)RC * Dm);
  _Float16* vc    = a16((size_t)RC * Dm);
  _Float16* ctxc  = a16((size_t)RC * Dm);
  float*    hb    = a32((size_t)RC * Dm);
  _Float16* hn    = a16((size_t)RC * Dm);
  _Float16* ffc   = a16((size_t)RC * FFd);
  _Float16* h2_h  = a16((size_t)RC * Dm);
  float*    x2    = a32((size_t)RM * Dm);
  _Float16* xn    = a16((size_t)RM * Dm);
  _Float16* qm    = a16((size_t)RM * Dm);
  _Float16* km    = a16((size_t)RM * Dm);
  _Float16* vm    = a16((size_t)RM * Dm);
  _Float16* ctxm  = a16((size_t)RM * Dm);
  float*    ab    = a32((size_t)RM * Dm);
  _Float16* an    = a16((size_t)RM * Dm);
  _Float16* ffm   = a16((size_t)RM * FFd);

  auto convT = [&](const float* s, _Float16* d, int K_, int N_) {
    transpose_f32_to_f16<<<dim3(N_ / 32, K_ / 32), 256, 0, stream>>>(s, d, K_, N_);
  };
  convT(cl_W,   clW_t,  CPD, Dm);
  convT(cWq,    cWq_t,  Dm,  Dm);  convT(cWk, cWk_t, Dm, Dm);
  convT(cWv,    cWv_t,  Dm,  Dm);  convT(cWo, cWo_t, Dm, Dm);
  convT(cff_W1, cW1_t,  Dm,  FFd); convT(cff_W2, cW2_t, FFd, Dm);
  convT(rev_W,  revW_t, Dm,  CPD);
  convT(Wq,     Wq_t,   Dm,  Dm);  convT(Wk, Wk_t, Dm, Dm);
  convT(Wv,     Wv_t,   Dm,  Dm);  convT(Wo, Wo_t, Dm, Dm);
  convT(ff_W1,  W1_t,   Dm,  FFd); convT(ff_W2, W2_t, FFd, Dm);

  const int gyC = (RC + 63) / 64;   // 22
  const int gyM = (RM + 63) / 64;   // 200
  const float qscale = 0.125f;      // 1/sqrt(64)

  // ---- clip path ----
  gather_clips<<<RC, 256, 0, stream>>>(x, flat);
  layernorm_to_h<<<RC, 256, 0, stream>>>(flat, cl_ln_g, cl_ln_b, flatn, CPD);
  gemm_kernel<5><<<dim3(Dm / 128, gyC), 256, 0, stream>>>(
      flatn, clW_t, cl_b, nullptr, cl, cl_h, RC, Dm, CPD, 0, 1.f);
  gemm_kernel<3><<<dim3(Dm / 128, gyC), 256, 0, stream>>>(
      cl_h, cWq_t, cbq, nullptr, nullptr, qc, RC, Dm, Dm, Cc, qscale);
  gemm_kernel<3><<<dim3(Dm / 128, gyC), 256, 0, stream>>>(
      cl_h, cWk_t, cbk, nullptr, nullptr, kc, RC, Dm, Dm, Cc, 1.f);
  gemm_kernel<3><<<dim3(Dm / 128, gyC), 256, 0, stream>>>(
      cl_h, cWv_t, cbv, nullptr, nullptr, vc, RC, Dm, Dm, Cc, 1.f);
  flash_attn<<<dim3(1, Bq * Hh), 256, 0, stream>>>(qc, kc, vc, ctxc, Cc);
  gemm_kernel<2><<<dim3(Dm / 128, gyC), 256, 0, stream>>>(
      ctxc, cWo_t, cbo, cl, hb, nullptr, RC, Dm, Dm, 0, 1.f);
  layernorm_to_h<<<RC, 256, 0, stream>>>(hb, cff_ln_g, cff_ln_b, hn, Dm);
  gemm_kernel<1><<<dim3(FFd / 128, gyC), 256, 0, stream>>>(
      hn, cW1_t, cff_b1, nullptr, nullptr, ffc, RC, FFd, Dm, 0, 1.f);
  gemm_kernel<6><<<dim3(Dm / 128, gyC), 256, 0, stream>>>(
      ffc, cW2_t, cff_b2, hb, hb, h2_h, RC, Dm, FFd, 0, 1.f);
  gemm_kernel<0><<<dim3(CPD / 128, gyC), 256, 0, stream>>>(   // rev -> reuse `flat`
      h2_h, revW_t, rev_b, nullptr, flat, nullptr, RC, CPD, Dm, 0, 1.f);
  combine_x2<<<((size_t)RM * Dm + 255) / 256, 256, 0, stream>>>(x, flat, x2);

  // ---- main path ----
  layernorm_to_h<<<RM, 256, 0, stream>>>(x2, ln_g, ln_b, xn, Dm);
  gemm_kernel<3><<<dim3(Dm / 128, gyM), 256, 0, stream>>>(
      xn, Wq_t, bqv, nullptr, nullptr, qm, RM, Dm, Dm, Mseq, qscale);
  gemm_kernel<3><<<dim3(Dm / 128, gyM), 256, 0, stream>>>(
      xn, Wk_t, bkv, nullptr, nullptr, km, RM, Dm, Dm, Mseq, 1.f);
  gemm_kernel<3><<<dim3(Dm / 128, gyM), 256, 0, stream>>>(
      xn, Wv_t, bvv, nullptr, nullptr, vm, RM, Dm, Dm, Mseq, 1.f);
  flash_attn<<<dim3((Mseq + 63) / 64, Bq * Hh), 256, 0, stream>>>(qm, km, vm, ctxm, Mseq);
  gemm_kernel<2><<<dim3(Dm / 128, gyM), 256, 0, stream>>>(
      ctxm, Wo_t, bov, x2, ab, nullptr, RM, Dm, Dm, 0, 1.f);
  layernorm_to_h<<<RM, 256, 0, stream>>>(ab, ff_ln_g, ff_ln_b, an, Dm);
  gemm_kernel<1><<<dim3(FFd / 128, gyM), 256, 0, stream>>>(
      an, W1_t, ff_b1, nullptr, nullptr, ffm, RM, FFd, Dm, 0, 1.f);
  gemm_kernel<2><<<dim3(Dm / 128, gyM), 256, 0, stream>>>(
      ffm, W2_t, ff_b2, ab, (float*)d_out, nullptr, RM, Dm, FFd, 0, 1.f);
}
